// Tracker_84018150244394
// MI455X (gfx1250) — matrix operand order repaired and checked by reading, not compiled, not run on hardware
//
#include <hip/hip_runtime.h>
#include <hip/hip_bf16.h>
#include <math.h>

#define Pp  32
#define Bn  64
#define Tn  100
#define Wn  50
#define En  64
#define Hn  256
#define G5  1280   // 5*H

typedef __attribute__((ext_vector_type(16))) __bf16         v16bf;
typedef __attribute__((ext_vector_type(16))) unsigned short v16us;
typedef __attribute__((ext_vector_type(8)))  float          v8f;

__device__ __forceinline__ unsigned short f2bf_rne(float f) {
  union { float f; unsigned int u; } x; x.f = f;
  unsigned int u = x.u;
  u += 0x7FFFu + ((u >> 16) & 1u);   // round-to-nearest-even
  return (unsigned short)(u >> 16);
}
__device__ __forceinline__ float sigf(float x) { return 1.0f / (1.0f + expf(-x)); }

// A-fragment position for h element (m, k) in the 16-bit A 16x32 WMMA layout:
//   lane = (m&15) + 16*hi,  hi=1 for k%32 in [8,16)|[24,32)
//   elem = (k&7) + 8*(k%32 >= 16)
// Buffer layout: [mt(2)][kt(8)][lane(32)][elem(16)]  (ushort)
__device__ __forceinline__ int afrag_pos(int m, int k) {
  int kt = k >> 5;
  int kk = k & 31;
  int hi = (kk >> 3) & 1;
  int elem = (kk & 7) + ((kk >> 4) << 3);
  int lane = (m & 15) + (hi << 4);
  int mt = m >> 4;
  return (((mt * 8 + kt) * 32) + lane) * 16 + elem;
}

// ---------------------------------------------------------------------------
// Kernel 1: pack W_hh (256x1280 f32, row-major [k][n]) into bf16 fragments in
// the V_WMMA_F32_16X16X32_BF16 B-matrix layout (K=32 x N=16 tile, 8 VGPRs):
//   lanes 0-15 : N=lane,    K = 0..7, 16..23
//   lanes16-31 : N=lane-16, K = 8..15, 24..31
// Tile order: [nt (80)][kt (8)][lane (32)][elem (16)]
// ---------------------------------------------------------------------------
__global__ void pack_whh(const float* __restrict__ W_hh,
                         unsigned short* __restrict__ out) {
  int g = blockIdx.x * blockDim.x + threadIdx.x;
  if (g >= 80 * 8 * 512) return;
  int nt   = g >> 12;
  int rem  = g & 4095;
  int kt   = rem >> 9;
  int q    = rem & 511;
  int lane = q >> 4;
  int pp   = q & 15;
  int koff = (lane < 16) ? 0 : 8;
  int k    = kt * 32 + koff + (pp & 7) + ((pp >> 3) << 4);
  int n    = nt * 16 + (lane & 15);
  out[g] = f2bf_rne(W_hh[k * G5 + n]);
}

// ---------------------------------------------------------------------------
// Kernel 2: per (t,b): emb = relu(prev_window @ W_act + b_act)  (E=64)
//   xw[t,b,:]  = emb @ W_ih + b_ih + b_hh          (1280)
//   xobs[t,b]  = emb @ W_obs[H:H+E] + b_obs        (scalar)
// ---------------------------------------------------------------------------
__global__ void precompute_x(const float* __restrict__ pw_in,
                             const float* __restrict__ W_act,
                             const float* __restrict__ b_act,
                             const float* __restrict__ W_ih,
                             const float* __restrict__ b_ih,
                             const float* __restrict__ b_hh,
                             const float* __restrict__ W_obs,
                             const float* __restrict__ b_obs,
                             float* __restrict__ xw,
                             float* __restrict__ xobs) {
  int blk = blockIdx.x;
  int t = blk / Bn, b = blk % Bn;
  int tid = threadIdx.x;
  __shared__ float spw[Wn];
  __shared__ float semb[En];
  if (tid < Wn) spw[tid] = pw_in[((size_t)b * Tn + t) * Wn + tid];
  __syncthreads();
  float s = b_act[tid];
  for (int w = 0; w < Wn; ++w) s += spw[w] * W_act[w * En + tid];
  semb[tid] = fmaxf(s, 0.0f);
  __syncthreads();
  for (int q = 0; q < 20; ++q) {
    int n = tid + 64 * q;
    float acc = b_ih[n] + b_hh[n];
    for (int e = 0; e < En; ++e) acc += semb[e] * W_ih[e * G5 + n];
    xw[((size_t)t * Bn + b) * G5 + n] = acc;
  }
  if (tid == 0) {
    float acc = b_obs[0];
    for (int e = 0; e < En; ++e) acc += semb[e] * W_obs[Hn + e];
    xobs[t * Bn + b] = acc;
  }
}

// ---------------------------------------------------------------------------
// Main persistent kernel: one workgroup per batch element b, 512 threads
// (16 waves). Wave j owns hidden-column tile j and its 5 gate N-tiles.
// h state lives in LDS as bf16 *pre-laid-out A fragments*; c state as f32.
// ---------------------------------------------------------------------------
__global__ __launch_bounds__(512, 1)
void smc_lstm(const float* __restrict__ h0, const float* __restrict__ c0,
              const float* __restrict__ eps, const float* __restrict__ gum,
              const unsigned short* __restrict__ whhp,
              const float* __restrict__ xw, const float* __restrict__ xobs,
              const float* __restrict__ W_obs, const float* __restrict__ W_lab,
              const float* __restrict__ b_lab,
              float* __restrict__ out) {
  __shared__ __attribute__((aligned(32))) unsigned short sA[2 * 8 * 32 * 16]; // 16 KB bf16 h fragments
  __shared__ float sC0[Pp * Hn];   // 32 KB current c
  __shared__ float sH1[Pp * Hn];   // 32 KB pre-resample h1 (f32)
  __shared__ float sC1[Pp * Hn];   // 32 KB pre-resample c1
  __shared__ float sLogpdf[Pp];
  __shared__ float sD1[Pp];
  __shared__ float sP[Pp];
  __shared__ int   sIdx[Pp];

  const int b    = blockIdx.x;
  const int tid  = threadIdx.x;
  const int wave = tid >> 5;
  const int lane = tid & 31;
  const int j    = wave;
  const int colg = j * 16 + (lane & 15);
  const float wobs_c = W_obs[colg];
  const float wlab_c = W_lab[colg];
  const float blab   = b_lab[0];

  // ---- init state: h0 -> bf16 A fragments, c0 -> f32 ----
  for (int i = 0; i < 16; ++i) {
    int e = tid + 512 * i;
    int m = e >> 8, col = e & 255;
    float hv = h0[((size_t)m * Bn + b) * Hn + col];
    sA[afrag_pos(m, col)] = f2bf_rne(hv);
    sC0[e] = c0[((size_t)m * Bn + b) * Hn + col];
  }
  if (tid < Pp) {
    sP[tid] = logf(1.0f / 32.0f);
    sLogpdf[tid] = 0.0f;
    sD1[tid] = 0.0f;
  }
  __syncthreads();

  for (int t = 0; t < Tn; ++t) {
    // ===== GEMM: gates(32x1280) = bias_row + h(32x256) @ W_hh(256x1280) =====
    v8f acc[5][2];
    const float* xwrow = xw + ((size_t)t * Bn + b) * G5;
#pragma unroll
    for (int g = 0; g < 5; ++g) {
      float bias = xwrow[g * Hn + colg];
      v8f v;
#pragma unroll
      for (int r = 0; r < 8; ++r) v[r] = bias;
      acc[g][0] = v;
      acc[g][1] = v;
    }

#pragma unroll 1
    for (int kt = 0; kt < 8; ++kt) {
      // A fragments: contiguous 32B per lane, already bf16 + laid out
      v16us a0u = *(const v16us*)&sA[((0 * 8 + kt) * 32 + lane) * 16];
      v16us a1u = *(const v16us*)&sA[((1 * 8 + kt) * 32 + lane) * 16];
      v16bf a0 = __builtin_bit_cast(v16bf, a0u);
      v16bf a1 = __builtin_bit_cast(v16bf, a1u);
#pragma unroll
      for (int g = 0; g < 5; ++g) {
        int nt = g * 16 + j;
        v16us bu = *(const v16us*)(whhp + ((size_t)(nt * 8 + kt) * 512 + lane * 16));
        v16bf bb = __builtin_bit_cast(v16bf, bu);
        acc[g][0] = __builtin_amdgcn_wmma_f32_16x16x32_bf16(
            false, a0, false, bb, (short)0, acc[g][0], false, false);
        acc[g][1] = __builtin_amdgcn_wmma_f32_16x16x32_bf16(
            false, a1, false, bb, (short)0, acc[g][1], false, false);
      }
    }

    // ===== elementwise LSTM + fused per-particle dot products =====
    const int mh = (lane >> 4) * 8;
#pragma unroll
    for (int mt = 0; mt < 2; ++mt) {
#pragma unroll
      for (int r = 0; r < 8; ++r) {
        int m = mt * 16 + mh + r;
        float iv = acc[0][mt][r], fv = acc[1][mt][r], gv = acc[2][mt][r];
        float ov = acc[3][mt][r], vv = acc[4][mt][r];
        float cold = sC0[m * Hn + colg];
        float mu = sigf(fv) * cold + sigf(iv) * tanhf(gv);
        float sp = (vv > 20.0f) ? vv : log1pf(expf(vv));
        float ep = __builtin_nontemporal_load(
            &eps[(size_t)t * (Pp * Bn * Hn) + (size_t)(m * Bn + b) * Hn + colg]);
        float c1 = mu + sp * ep;
        float h1 = sigf(ov) * tanhf(c1);
        sH1[m * Hn + colg] = h1;
        sC1[m * Hn + colg] = c1;
        float lp = h1 * wobs_c;
        float dl = h1 * wlab_c;
#pragma unroll
        for (int msk = 1; msk <= 8; msk <<= 1) {
          lp += __shfl_xor(lp, msk, 32);
          dl += __shfl_xor(dl, msk, 32);
        }
        if ((lane & 15) == 0) {
          atomicAdd(&sLogpdf[m], lp);
          atomicAdd(&sD1[m], dl);
        }
      }
    }
    __syncthreads();   // (A) h1/c1/logpdf/d1 complete

    // ===== particle weights, Gumbel-max resampling, outputs (wave 0) =====
    if (wave == 0) {
      float lw = sP[lane] + sLogpdf[lane] + xobs[t * Bn + b];
      float mx = lw;
#pragma unroll
      for (int msk = 1; msk <= 16; msk <<= 1) mx = fmaxf(mx, __shfl_xor(mx, msk, 32));
      float ex = expf(lw - mx), sm = ex;
#pragma unroll
      for (int msk = 1; msk <= 16; msk <<= 1) sm += __shfl_xor(sm, msk, 32);
      float p1 = lw - mx - logf(sm);
      float w1 = expf(p1);
      float logit = logf(0.5f * w1 + 0.015625f);   // ALPHA=0.5, (1-A)/P

      const float* grow = gum + ((size_t)(t * Bn + b)) * (Pp * Pp) + lane * Pp;
      float best = -3.0e38f; int bi = 0;
      for (int k = 0; k < Pp; ++k) {
        float v = __shfl(logit, k, 32) + __builtin_nontemporal_load(&grow[k]);
        if (v > best) { best = v; bi = k; }
      }
      sIdx[lane] = bi;

      float wsel = __shfl(w1, bi, 32);
      wsel = wsel / (0.5f * wsel + 0.015625f);
      float lg = logf(wsel);
      float mx2 = lg;
#pragma unroll
      for (int msk = 1; msk <= 16; msk <<= 1) mx2 = fmaxf(mx2, __shfl_xor(mx2, msk, 32));
      float ex2 = expf(lg - mx2), sm2 = ex2;
#pragma unroll
      for (int msk = 1; msk <= 16; msk <<= 1) sm2 += __shfl_xor(sm2, msk, 32);
      float pn = lg - mx2 - logf(sm2);
      sP[lane] = pn;

      float d1i = sD1[bi];
      out[(size_t)Tn * Bn + (size_t)t * (Pp * Bn) + lane * Bn + b] = sigf(d1i + blab);
      float contrib = expf(pn) * d1i;
#pragma unroll
      for (int msk = 1; msk <= 16; msk <<= 1) contrib += __shfl_xor(contrib, msk, 32);
      if (lane == 0) out[t * Bn + b] = sigf(contrib + blab);

      sLogpdf[lane] = 0.0f;
      sD1[lane] = 0.0f;
    }
    __syncthreads();   // (B) idx ready, accumulators zeroed

    // ===== resample gather: bf16-pack h_new into A fragments; c_new f32 =====
    for (int i = 0; i < 16; ++i) {
      int e = tid + 512 * i;
      int m = e >> 8, col = e & 255;
      int src = sIdx[m];
      float hv = sH1[src * Hn + col];
      sA[afrag_pos(m, col)] = f2bf_rne(hv);
      sC0[e] = sC1[src * Hn + col];
    }
    __syncthreads();   // (C) state ready for next GEMM
  }
}

// ---------------------------------------------------------------------------
extern "C" void kernel_launch(void* const* d_in, const int* in_sizes, int n_in,
                              void* d_out, int out_size, void* d_ws, size_t ws_size,
                              hipStream_t stream) {
  const float* prev_window = (const float*)d_in[0];
  const float* h0    = (const float*)d_in[1];
  const float* c0    = (const float*)d_in[2];
  const float* eps   = (const float*)d_in[3];
  const float* gum   = (const float*)d_in[4];
  const float* W_act = (const float*)d_in[5];
  const float* b_act = (const float*)d_in[6];
  const float* W_ih  = (const float*)d_in[7];
  const float* b_ih  = (const float*)d_in[8];
  const float* W_hh  = (const float*)d_in[9];
  const float* b_hh  = (const float*)d_in[10];
  const float* W_obs = (const float*)d_in[11];
  const float* b_obs = (const float*)d_in[12];
  const float* W_lab = (const float*)d_in[13];
  const float* b_lab = (const float*)d_in[14];

  char* ws = (char*)d_ws;
  unsigned short* whhp = (unsigned short*)ws;                    //   655,360 B
  float* xw   = (float*)(ws + 655360);                           // 32,768,000 B
  float* xobs = (float*)(ws + 655360 + 32768000);                //     25,600 B

  pack_whh<<<1280, 256, 0, stream>>>(W_hh, whhp);
  precompute_x<<<Tn * Bn, 64, 0, stream>>>(prev_window, W_act, b_act, W_ih,
                                           b_ih, b_hh, W_obs, b_obs, xw, xobs);
  smc_lstm<<<Bn, 512, 0, stream>>>(h0, c0, eps, gum, whhp, xw, xobs,
                                   W_obs, W_lab, b_lab, (float*)d_out);
}
